// Transformer_15255723835402
// MI455X (gfx1250) — compile-verified, matrix operand order checked by reading, not run on hardware
//
#include <hip/hip_runtime.h>
#include <hip/hip_bf16.h>
#include <math.h>
#include <stdint.h>

typedef _Float16 f16;
typedef __attribute__((ext_vector_type(16))) _Float16 v16h;
typedef __attribute__((ext_vector_type(8)))  float    v8f;

#define B_    32
#define N_    256
#define DIM_  384
#define H_    6
#define DH_   64
#define MLP_  1536
#define DEPTH_ 12
#define ROWS_ (B_ * N_)   // 8192

// ---------------------------------------------------------------------------
// CDNA5 WMMA fragment helpers (ISA 7.12.2, wave32).
// A run of 8 consecutive K values (one 16-byte octet) lands CONTIGUOUSLY in
// one lane's fragment storage: octet o of a 32-wide K block maps to
//   lane += (o&1)*16 , byte offset (o>>1)*16.
// So K-contiguous global data can be staged into swizzled LDS with per-lane
// 16-byte async copies (global_load_async_to_lds_b128).
// ---------------------------------------------------------------------------
__device__ __forceinline__ int fq_idx(int k) {   // uint slot 0..7, k even
  return ((k >= 16) ? 4 : 0) + ((k & 7) >> 1);
}
__device__ __forceinline__ int fl_off(int k) {   // lane-half offset 0/16
  return ((k & 15) >= 8) ? 16 : 0;
}
__device__ __forceinline__ v8f wmma16(v16h a, v16h b, v8f c) {
  return __builtin_amdgcn_wmma_f32_16x16x32_f16(false, a, false, b,
                                                (short)0, c, false, false);
}
// LDS byte offset of chunk c (row = c>>2 within 128-row/col tile, o = c&3)
__device__ __forceinline__ int chunk_lds(int c) {
  const int row = c >> 2, o = c & 3;
  return (row >> 4) * 1024 + ((row & 15) + ((o & 1) << 4)) * 32 + ((o >> 1) << 4);
}
// ---- CDNA5 async global->LDS copy (ASYNCcnt tracked), via inline asm ------
__device__ __forceinline__ void async_b128(unsigned lds_addr, const void* gptr) {
  asm volatile("global_load_async_to_lds_b128 %0, %1, off"
               :: "v"(lds_addr), "v"((unsigned long long)(uintptr_t)gptr)
               : "memory");
}
__device__ __forceinline__ void wait_async0() {
  asm volatile("s_wait_asynccnt 0x0" ::: "memory");
}
__device__ __forceinline__ void wait_async4() {
  asm volatile("s_wait_asynccnt 0x4" ::: "memory");
}
union U4 { uint4 v; unsigned short h[8]; unsigned int w[4]; };

// ---------------------------------------------------------------------------
// fp32 -> f16 weight conversion WITH transpose: s[L][K][N] -> d[L][N][K].
// Makes the GEMM B operand K-contiguous so it can be async-staged.
// ---------------------------------------------------------------------------
__global__ void f32_to_f16_T(const float* __restrict__ s, f16* __restrict__ d,
                             int K, int N) {
  const int KN = K * N;
  for (int i = blockIdx.x * blockDim.x + threadIdx.x; i < DEPTH_ * KN;
       i += gridDim.x * blockDim.x) {
    int L = i / KN, rem = i % KN;
    int r = rem / N, c = rem % N;
    d[(size_t)L * KN + (size_t)c * K + r] = (f16)s[i];
  }
}

// ---------------------------------------------------------------------------
// LayerNorm (fp32 in) -> f16 out.  One 128-thread block per row.
// ---------------------------------------------------------------------------
__global__ __launch_bounds__(128)
void layernorm_f16(const float* __restrict__ x, const float* __restrict__ g,
                   const float* __restrict__ b, f16* __restrict__ y) {
  __shared__ float red[128];
  const int row = blockIdx.x;
  const float* xr = x + (size_t)row * DIM_;
  float s = 0.f;
  for (int c = threadIdx.x; c < DIM_; c += 128) s += xr[c];
  red[threadIdx.x] = s; __syncthreads();
  for (int o = 64; o > 0; o >>= 1) {
    if (threadIdx.x < o) red[threadIdx.x] += red[threadIdx.x + o];
    __syncthreads();
  }
  const float mu = red[0] * (1.f / DIM_);
  __syncthreads();
  float v = 0.f;
  for (int c = threadIdx.x; c < DIM_; c += 128) { float d = xr[c] - mu; v += d * d; }
  red[threadIdx.x] = v; __syncthreads();
  for (int o = 64; o > 0; o >>= 1) {
    if (threadIdx.x < o) red[threadIdx.x] += red[threadIdx.x + o];
    __syncthreads();
  }
  const float rstd = rsqrtf(red[0] * (1.f / DIM_) + 1e-5f);
  for (int c = threadIdx.x; c < DIM_; c += 128)
    y[(size_t)row * DIM_ + c] = (f16)(((xr[c] - mu) * rstd) * g[c] + b[c]);
}

// ---------------------------------------------------------------------------
// f16 WMMA GEMM: C[M,N] = A[M,K] * Bt[N,K]^T (f32 accum).
// Block tile 128x128, BK=32; 8 waves (4x2), each wave owns a 32x64 C tile.
// Both operands are K-contiguous in memory -> staged with double-buffered
// global_load_async_to_lds_b128 (4 async wave-instructions per stage,
// s_wait_asynccnt 4/0 pipeline).
// mode 0: store f16.  mode 1: +bias, exact GELU, store f16.
// mode 2: (acc + bias) * ls accumulated into fp32 Of (LayerScale+residual).
// ---------------------------------------------------------------------------
__global__ __launch_bounds__(256, 1)
void gemm_wmma(const f16* __restrict__ A, const f16* __restrict__ Bt,
               const float* __restrict__ bias, const float* __restrict__ ls,
               f16* __restrict__ Oh, float* __restrict__ Of,
               int M, int N, int K, int mode) {
  __shared__ __align__(32) f16 As[2][8 * 512];   // double-buffered A frags
  __shared__ __align__(32) f16 Bs[2][8 * 512];   // double-buffered B frags
  const int tid = threadIdx.x, lane = tid & 31, w = tid >> 5;
  const int lhalf = lane >> 4, cn = lane & 15;
  const int bm = blockIdx.y * 128, bn = blockIdx.x * 128;
  const int wm = w & 3, wn = w >> 2;
  const v8f vz = {};
  v8f acc[2][4];
#pragma unroll
  for (int i = 0; i < 2; ++i)
#pragma unroll
    for (int j = 0; j < 4; ++j) acc[i][j] = vz;

  // Each thread stages 2 A-chunks and 2 B-chunks (16B each) per K-step.
  const int c0 = tid * 2, c1 = tid * 2 + 1;
  const int lo0 = chunk_lds(c0), lo1 = chunk_lds(c1);
  const unsigned aBase = (unsigned)(uintptr_t)&As[0][0];
  const unsigned bBase = (unsigned)(uintptr_t)&Bs[0][0];
  const size_t ga0 = (size_t)(bm + (c0 >> 2)) * K + (c0 & 3) * 8;
  const size_t ga1 = (size_t)(bm + (c1 >> 2)) * K + (c1 & 3) * 8;
  const size_t gb0 = (size_t)(bn + (c0 >> 2)) * K + (c0 & 3) * 8;
  const size_t gb1 = (size_t)(bn + (c1 >> 2)) * K + (c1 & 3) * 8;

  auto issue = [&](int p, int k0) {
    async_b128(aBase + p * 8192 + lo0, A + ga0 + k0);
    async_b128(aBase + p * 8192 + lo1, A + ga1 + k0);
    async_b128(bBase + p * 8192 + lo0, Bt + gb0 + k0);
    async_b128(bBase + p * 8192 + lo1, Bt + gb1 + k0);
  };

  const int nk = K >> 5;
  issue(0, 0);
  for (int i = 0; i < nk; ++i) {
    const int p = i & 1;
    if (i + 1 < nk) { issue(p ^ 1, (i + 1) << 5); wait_async4(); }
    else            { wait_async0(); }
    __syncthreads();                 // stage p visible to all waves
    v16h af[2], bf[4];
#pragma unroll
    for (int mf = 0; mf < 2; ++mf)
      af[mf] = *(const v16h*)&As[p][(2 * wm + mf) * 512 + lane * 16];
#pragma unroll
    for (int nf = 0; nf < 4; ++nf)
      bf[nf] = *(const v16h*)&Bs[p][(4 * wn + nf) * 512 + lane * 16];
#pragma unroll
    for (int mf = 0; mf < 2; ++mf)
#pragma unroll
      for (int nf = 0; nf < 4; ++nf)
        acc[mf][nf] = wmma16(af[mf], bf[nf], acc[mf][nf]);
    __syncthreads();                 // done reading p before it is refilled
  }

#pragma unroll
  for (int mf = 0; mf < 2; ++mf)
#pragma unroll
    for (int nf = 0; nf < 4; ++nf) {
      const int gcol = bn + wn * 64 + nf * 16 + cn;
#pragma unroll
      for (int r = 0; r < 8; ++r) {
        const int grow = bm + wm * 32 + mf * 16 + r + lhalf * 8;
        const float v = acc[mf][nf][r];
        if (mode == 0) {
          Oh[(size_t)grow * N + gcol] = (f16)v;
        } else if (mode == 1) {
          float t = v + bias[gcol];
          t = 0.5f * t * (1.f + erff(t * 0.70710678118f));  // exact GELU
          Oh[(size_t)grow * N + gcol] = (f16)t;
        } else {
          Of[(size_t)grow * N + gcol] += (v + bias[gcol]) * ls[gcol];
        }
      }
    }
}

// ---------------------------------------------------------------------------
// Fused talking-heads attention.  Grid (N/16, B); 256 threads (8 waves).
// Q and K^T tiles are K-contiguous in memory -> async-staged into swizzled
// LDS.  Pass 1: per-head dots (WMMA) -> scale -> pre-softmax head mix; wave g
// keeps the full mixed 16x256 score row in registers.  Diagonal mask.
// Softmax: register + __shfl_xor reductions (16-lane column groups, wave32).
// Pass 2: post-softmax head mix -> A-fragments in LDS -> P' @ V via WMMA.
// ---------------------------------------------------------------------------
__global__ __launch_bounds__(256, 1)
void attn_wmma(const f16* __restrict__ q, const f16* __restrict__ kv,
               const float* __restrict__ scale_p, const float* __restrict__ mixp_p,
               const float* __restrict__ mixq_p, f16* __restrict__ o_out) {
  __shared__ __align__(32) f16  QA[12 * 512];  // Q A-frags [h][kf]
  __shared__ __align__(32) f16  KV[24 * 512];  // K^T / V B-frags per chunk
  __shared__ float SP[6 * 512];                // per-head scores / probs (plain)
  __shared__ __align__(32) f16  PA[6 * 512];   // mixed-prob A-frags per chunk
  __shared__ float MP[H_ * H_], MQ[H_ * H_], SC[H_];
  unsigned int* const KV32 = (unsigned int*)KV;
  unsigned int* const PA32 = (unsigned int*)PA;

  const int tid = threadIdx.x, lane = tid & 31, w = tid >> 5;
  const int lhalf = lane >> 4, cn = lane & 15;
  const int bb = blockIdx.y, i0 = blockIdx.x * 16;
  const size_t qbase = (size_t)bb * N_ * DIM_;
  const size_t kvbase = (size_t)bb * N_ * (2 * DIM_);
  const unsigned qaBase = (unsigned)(uintptr_t)&QA[0];
  const unsigned kvBase = (unsigned)(uintptr_t)&KV[0];

  if (tid < 36)        MP[tid] = mixp_p[tid];
  else if (tid < 72)   MQ[tid - 36] = mixq_p[tid - 36];
  else if (tid < 78)   SC[tid - 72] = scale_p[tid - 72];

  // Async-stage Q tile (16 x 384): 768 16-byte octets, 3 per thread.
  for (int c = tid; c < 768; c += 256) {
    const int row = c / 48, oc = c % 48, o = oc & 3;
    const unsigned lds = qaBase + (oc >> 2) * 1024 +
                         (row + ((o & 1) << 4)) * 32 + ((o >> 1) << 4);
    async_b128(lds, q + qbase + (size_t)(i0 + row) * DIM_ + oc * 8);
  }

  float smix[16][8];            // mixed scores: 16 frags (8 chunks x 2) x 8 rows
  const v8f vz = {};
  v8f oacc[3];
  oacc[0] = vz; oacc[1] = vz; oacc[2] = vz;

  // ---- pass 1: scores ----
#pragma unroll
  for (int jc = 0; jc < 8; ++jc) {
    __syncthreads();              // KV free, prior SP reads done
    const int j0 = jc * 32;
    // Async-stage K^T chunk: 1536 octets (32 keys x 48), 6 per thread.
    // Fragment id = (h*2+kf)*2 + (j>>4); lane = (j&15) + octet parity.
    for (int c = tid; c < 1536; c += 256) {
      const int j = c / 48, oc = c % 48, o = oc & 3;
      const unsigned lds = kvBase + ((oc >> 2) * 2 + (j >> 4)) * 1024 +
                           ((j & 15) + ((o & 1) << 4)) * 32 + ((o >> 1) << 4);
      async_b128(lds, kv + kvbase + (size_t)(j0 + j) * (2 * DIM_) + oc * 8);
    }
    wait_async0();
    __syncthreads();
    if (w < H_) {                          // wave h computes head-h dots
      const int h = w;
      const float sc = SC[h];
#pragma unroll
      for (int nf = 0; nf < 2; ++nf) {
        v8f cfr = vz;
#pragma unroll
        for (int kf = 0; kf < 2; ++kf) {
          v16h a  = *(const v16h*)&QA[(h * 2 + kf) * 512 + lane * 16];
          v16h bk = *(const v16h*)&KV[((h * 2 + kf) * 2 + nf) * 512 + lane * 16];
          cfr = wmma16(a, bk, cfr);
        }
#pragma unroll
        for (int r = 0; r < 8; ++r)
          SP[h * 512 + (r + lhalf * 8) * 32 + nf * 16 + cn] = cfr[r] * sc;
      }
    }
    __syncthreads();
    if (w < H_) {                          // wave g accumulates pre-mix
      const int g = w;
#pragma unroll
      for (int nf = 0; nf < 2; ++nf)
#pragma unroll
        for (int r = 0; r < 8; ++r) {
          const int i = r + lhalf * 8, jl = nf * 16 + cn;
          float a = 0.f;
#pragma unroll
          for (int h = 0; h < H_; ++h) a += MP[h * H_ + g] * SP[h * 512 + i * 32 + jl];
          if (i0 + i == j0 + jl) a = -1e30f;   // mask self position
          smix[jc * 2 + nf][r] = a;
        }
    }
  }

  // ---- softmax over j (row-wise across 16-lane column groups) ----
  if (w < H_) {
    float mx[8], sm[8];
#pragma unroll
    for (int r = 0; r < 8; ++r) mx[r] = -3.0e38f;
#pragma unroll
    for (int f = 0; f < 16; ++f)
#pragma unroll
      for (int r = 0; r < 8; ++r) mx[r] = fmaxf(mx[r], smix[f][r]);
#pragma unroll
    for (int off = 8; off > 0; off >>= 1)
#pragma unroll
      for (int r = 0; r < 8; ++r) mx[r] = fmaxf(mx[r], __shfl_xor(mx[r], off, 32));
#pragma unroll
    for (int r = 0; r < 8; ++r) sm[r] = 0.f;
#pragma unroll
    for (int f = 0; f < 16; ++f)
#pragma unroll
      for (int r = 0; r < 8; ++r) {
        float p = __expf(smix[f][r] - mx[r]);
        smix[f][r] = p; sm[r] += p;
      }
#pragma unroll
    for (int off = 8; off > 0; off >>= 1)
#pragma unroll
      for (int r = 0; r < 8; ++r) sm[r] += __shfl_xor(sm[r], off, 32);
#pragma unroll
    for (int r = 0; r < 8; ++r) {
      const float inv = 1.f / sm[r];
#pragma unroll
      for (int f = 0; f < 16; ++f) smix[f][r] *= inv;
    }
  }

  // ---- pass 2: post-mix + P' @ V ----
#pragma unroll
  for (int jc = 0; jc < 8; ++jc) {
    __syncthreads();
    const int j0 = jc * 32;
    if (w < H_) {                          // spill this chunk's probs (plain)
      const int g = w;
#pragma unroll
      for (int nf = 0; nf < 2; ++nf)
#pragma unroll
        for (int r = 0; r < 8; ++r)
          SP[g * 512 + (r + lhalf * 8) * 32 + nf * 16 + cn] = smix[jc * 2 + nf][r];
    }
    // Stage V chunk as B-fragments: B[kk=j][n=d]; packed j-pairs (two rows).
    for (int u = tid; u < 16 * 48; u += 256) {
      int jp = u & 15, cg = u >> 4;
      int j = jp * 2, c0 = cg * 8;
      const f16* vsrc = kv + kvbase + (size_t)(j0 + j) * (2 * DIM_) + DIM_ + c0;
      U4 r0, r1;
      r0.v = *(const uint4*)vsrc;
      r1.v = *(const uint4*)(vsrc + 2 * DIM_);
      const int tq = fq_idx(j), loff = fl_off(j);
#pragma unroll
      for (int u2 = 0; u2 < 8; ++u2) {
        int c = c0 + u2;
        int g = c >> 6, d = c & 63, nf = d >> 4, nn = d & 15;
        unsigned int val = (unsigned int)r0.h[u2] | ((unsigned int)r1.h[u2] << 16);
        KV32[(g * 4 + nf) * 256 + (nn + loff) * 8 + tq] = val;
      }
    }
    __syncthreads();
    // Post-softmax head mix -> A-fragment layout; packed jl-pairs.
    for (int e = tid; e < 6 * 16 * 16; e += 256) {
      int g = e >> 8, rem = e & 255, i = rem >> 4, jl = (rem & 15) * 2;
      float a0 = 0.f, a1 = 0.f;
#pragma unroll
      for (int h = 0; h < H_; ++h) {
        const float m = MQ[h * H_ + g];
        a0 += m * SP[h * 512 + i * 32 + jl];
        a1 += m * SP[h * 512 + i * 32 + jl + 1];
      }
      f16 two[2] = {(f16)a0, (f16)a1};
      PA32[g * 256 + (i + fl_off(jl)) * 8 + fq_idx(jl)] = *(const unsigned int*)two;
    }
    __syncthreads();
    // o accumulation: 24 (g,nf) tile-units over 8 waves, 3 each
#pragma unroll
    for (int z = 0; z < 3; ++z) {
      const int u = w * 3 + z, g = u >> 2, nf = u & 3;
      v16h pa = *(const v16h*)&PA[g * 512 + lane * 16];
      v16h vb = *(const v16h*)&KV[(g * 4 + nf) * 512 + lane * 16];
      oacc[z] = wmma16(pa, vb, oacc[z]);
    }
  }

  // Store o tiles (f16, row-major [b,n,INNER])
#pragma unroll
  for (int z = 0; z < 3; ++z) {
    const int u = w * 3 + z, g = u >> 2, nf = u & 3;
#pragma unroll
    for (int r = 0; r < 8; ++r) {
      const int i = r + lhalf * 8;
      o_out[((size_t)bb * N_ + i0 + i) * DIM_ + g * 64 + nf * 16 + cn] =
          (f16)oacc[z][r];
    }
  }
}

// ---------------------------------------------------------------------------
extern "C" void kernel_launch(void* const* d_in, const int* in_sizes, int n_in,
                              void* d_out, int out_size, void* d_ws, size_t ws_size,
                              hipStream_t stream) {
  (void)in_sizes; (void)n_in; (void)out_size; (void)ws_size;
  const float* x    = (const float*)d_in[0];
  const float* ln1g = (const float*)d_in[1];
  const float* ln1b = (const float*)d_in[2];
  const float* wq   = (const float*)d_in[3];
  const float* wkv  = (const float*)d_in[4];
  const float* sc   = (const float*)d_in[5];
  const float* mp   = (const float*)d_in[6];
  const float* mq   = (const float*)d_in[7];
  const float* wo   = (const float*)d_in[8];
  const float* bo   = (const float*)d_in[9];
  const float* ls1  = (const float*)d_in[10];
  const float* ln2g = (const float*)d_in[11];
  const float* ln2b = (const float*)d_in[12];
  const float* w1   = (const float*)d_in[13];
  const float* b1   = (const float*)d_in[14];
  const float* w2   = (const float*)d_in[15];
  const float* b2   = (const float*)d_in[16];
  const float* ls2  = (const float*)d_in[17];

  char* base = (char*)d_ws;
  size_t off = 0;
  auto take = [&](size_t elems) -> f16* {
    f16* p = (f16*)(base + off);
    off += ((elems * sizeof(f16)) + 255) & ~(size_t)255;
    return p;
  };
  f16* wqh  = take((size_t)DEPTH_ * DIM_ * DIM_);        // transposed [N][K]
  f16* wkvh = take((size_t)DEPTH_ * DIM_ * 2 * DIM_);    // transposed
  f16* woh  = take((size_t)DEPTH_ * DIM_ * DIM_);        // transposed
  f16* w1h  = take((size_t)DEPTH_ * DIM_ * MLP_);        // transposed
  f16* w2h  = take((size_t)DEPTH_ * MLP_ * DIM_);        // transposed
  f16* yh   = take((size_t)ROWS_ * DIM_);
  f16* qb   = take((size_t)ROWS_ * DIM_);
  f16* kvb  = take((size_t)ROWS_ * 2 * DIM_);
  f16* aob  = take((size_t)ROWS_ * DIM_);
  f16* hb   = take((size_t)ROWS_ * MLP_);

  // Convert + transpose all weights fp32 -> f16 [N][K] (deterministic).
  f32_to_f16_T<<<1024, 256, 0, stream>>>(wq,  wqh,  DIM_, DIM_);
  f32_to_f16_T<<<1024, 256, 0, stream>>>(wkv, wkvh, DIM_, 2 * DIM_);
  f32_to_f16_T<<<1024, 256, 0, stream>>>(wo,  woh,  DIM_, DIM_);
  f32_to_f16_T<<<1024, 256, 0, stream>>>(w1,  w1h,  DIM_, MLP_);
  f32_to_f16_T<<<1024, 256, 0, stream>>>(w2,  w2h,  MLP_, DIM_);

  // Running fp32 residual lives in d_out.
  float* xo = (float*)d_out;
  hipMemcpyAsync(xo, x, (size_t)ROWS_ * DIM_ * sizeof(float),
                 hipMemcpyDeviceToDevice, stream);

  for (int L = 0; L < DEPTH_; ++L) {
    // --- attention ---
    layernorm_f16<<<ROWS_, 128, 0, stream>>>(xo, ln1g + L * DIM_, ln1b + L * DIM_, yh);
    gemm_wmma<<<dim3(DIM_ / 128, ROWS_ / 128), 256, 0, stream>>>(
        yh, wqh + (size_t)L * DIM_ * DIM_, nullptr, nullptr, qb, nullptr,
        ROWS_, DIM_, DIM_, 0);
    gemm_wmma<<<dim3(2 * DIM_ / 128, ROWS_ / 128), 256, 0, stream>>>(
        yh, wkvh + (size_t)L * DIM_ * 2 * DIM_, nullptr, nullptr, kvb, nullptr,
        ROWS_, 2 * DIM_, DIM_, 0);
    attn_wmma<<<dim3(N_ / 16, B_), 256, 0, stream>>>(
        qb, kvb, sc + L * H_, mp + L * H_ * H_, mq + L * H_ * H_, aob);
    gemm_wmma<<<dim3(DIM_ / 128, ROWS_ / 128), 256, 0, stream>>>(
        aob, woh + (size_t)L * DIM_ * DIM_, bo + L * DIM_, ls1 + L * DIM_,
        nullptr, xo, ROWS_, DIM_, DIM_, 2);
    // --- FFN ---
    layernorm_f16<<<ROWS_, 128, 0, stream>>>(xo, ln2g + L * DIM_, ln2b + L * DIM_, yh);
    gemm_wmma<<<dim3(MLP_ / 128, ROWS_ / 128), 256, 0, stream>>>(
        yh, w1h + (size_t)L * DIM_ * MLP_, b1 + L * MLP_, nullptr, hb, nullptr,
        ROWS_, MLP_, DIM_, 1);
    gemm_wmma<<<dim3(DIM_ / 128, ROWS_ / 128), 256, 0, stream>>>(
        hb, w2h + (size_t)L * MLP_ * DIM_, b2 + L * DIM_, ls2 + L * DIM_,
        nullptr, xo, ROWS_, DIM_, MLP_, 2);
  }
}
